// transformerencoder_73418170958181
// MI455X (gfx1250) — compile-verified, matrix operand order checked by reading, not run on hardware
//
#include <hip/hip_runtime.h>

typedef __attribute__((ext_vector_type(16))) _Float16 v16h;
typedef __attribute__((ext_vector_type(8)))  _Float16 v8h;
typedef __attribute__((ext_vector_type(8)))  float    v8f;

// ---------------------------------------------------------------- device utils
__device__ __forceinline__ v8f wmma_f16f32(v16h a, v16h b, v8f c) {
#if defined(__HIP_DEVICE_COMPILE__)
  return __builtin_amdgcn_wmma_f32_16x16x32_f16(false, a, false, b, (short)0, c,
                                                false, false);
#else
  return c;
#endif
}

__device__ __forceinline__ void lds_fence() {
#if defined(__HIP_DEVICE_COMPILE__)
  asm volatile("s_wait_dscnt 0" ::: "memory");
#endif
}

// Load one lane's 16 f16 elements of a 16x32 A-fragment (or mirrored B-fragment).
// Caller passes p = base + row*ld + k0 + ((lane&16)?8:0).
// lane<16 : elems 0..7 = K k0..k0+7,    elems 8..15 = K k0+16..k0+23
// lane>=16: elems 0..7 = K k0+8..k0+15, elems 8..15 = K k0+24..k0+31
__device__ __forceinline__ v16h frag_load(const _Float16* p) {
  v8h lo = *(const v8h*)(p);
  v8h hi = *(const v8h*)(p + 16);
  v16h r;
#pragma unroll
  for (int i = 0; i < 8; ++i) { r[i] = lo[i]; r[i + 8] = hi[i]; }
  return r;
}

// ---------------------------------------------------------------- generic GEMM
// C[m,n] = act( sum_k A[m+joff, k] * Bt[n, k] + bias[n] + (accsrc ? accsrc[m,n] : 0) )
// One wave = one 16x64 tile (1 A-frag x 4 B-frags -> 4 WMMA per k-step, 4x A reuse).
// All 5 fragment loads of a k-step are issued before the first WMMA so they get
// distinct registers -> partial s_wait_loadcnt, loads overlap WMMA execution.
// M = 8192 tokens always.  K % 32 == 0.  Out-of-range lanes (conv row shift,
// N tail) read from a zeroed scratch page so all fragment loads are unconditional.
// omode 0: row-major [M,N]   omode 1: f16 out at [b,h,t,d]   omode 2: f16 out at [b,h,d,t]
__global__ void __launch_bounds__(256) k_gemm(
    const _Float16* __restrict__ A, int lda, int joff,
    const _Float16* __restrict__ Bt, int ldb,
    const float* __restrict__ bias, const float* __restrict__ accsrc,
    float* __restrict__ outF, _Float16* __restrict__ outH,
    const _Float16* __restrict__ zbuf,
    int M, int N, int K, int act, int omode) {
  int tilesN = (N + 63) >> 6;           // 64-wide N blocks
  int tiles = (M >> 4) * tilesN;
  int wave = blockIdx.x * 8 + (threadIdx.x >> 5);
  if (wave >= tiles) return;
  int lane = threadIdx.x & 31;
  int tm = wave / tilesN, tn = wave - tm * tilesN;
  int m0 = tm << 4, n0 = tn << 6;
  int lrw = lane & 15;
  int khi = (lane & 16) ? 8 : 0;
  int rb = m0 + ((lane & 16) ? 8 : 0);
  int arow = m0 + lrw;
  bool avalid = ((arow & 1023) + joff) >= 0;  // causal-conv shift stays in batch

  // loop-invariant per-lane base pointers; invalid lanes walk the zero page
  const _Float16* pa = avalid ? (A + (size_t)(arow + joff) * lda + khi) : (zbuf + khi);
  const _Float16* pb[4];
  int coln[4];
  bool bval[4];
#pragma unroll
  for (int nn = 0; nn < 4; ++nn) {
    coln[nn] = n0 + nn * 16 + lrw;
    bval[nn] = coln[nn] < N;
    pb[nn] = bval[nn] ? (Bt + (size_t)coln[nn] * ldb + khi) : (zbuf + khi);
  }

  v8f acc[4] = {};
  if (accsrc != nullptr) {
#pragma unroll
    for (int nn = 0; nn < 4; ++nn)
      if (bval[nn]) {
#pragma unroll
        for (int r = 0; r < 8; ++r)
          acc[nn][r] = accsrc[(size_t)(rb + r) * N + coln[nn]];
      }
  }
  for (int k0 = 0; k0 < K; k0 += 32) {
    // issue all loads first: distinct live values -> distinct regs -> partial waits
    v16h a  = frag_load(pa + k0);
    v16h b0 = frag_load(pb[0] + k0);
    v16h b1 = frag_load(pb[1] + k0);
    v16h b2 = frag_load(pb[2] + k0);
    v16h b3 = frag_load(pb[3] + k0);
    acc[0] = wmma_f16f32(a, b0, acc[0]);
    acc[1] = wmma_f16f32(a, b1, acc[1]);
    acc[2] = wmma_f16f32(a, b2, acc[2]);
    acc[3] = wmma_f16f32(a, b3, acc[3]);
  }
#pragma unroll
  for (int nn = 0; nn < 4; ++nn) {
    if (!bval[nn]) continue;
    int col = coln[nn];
    float bvs = (bias != nullptr) ? bias[col] : 0.0f;
#pragma unroll
    for (int r = 0; r < 8; ++r) {
      float v = acc[nn][r] + bvs;
      if (act == 1) v = (v >= 0.0f) ? v : 0.2f * v;       // leaky relu
      else if (act == 2) v = (v >= 0.0f) ? v : 0.0f;      // relu
      int row = rb + r;
      if (outF != nullptr) outF[(size_t)row * N + col] = v;
      if (outH != nullptr) {
        size_t idx;
        if (omode == 0) {
          idx = (size_t)row * N + col;
        } else {
          int bb = row >> 10, t = row & 1023, hh = col >> 6, d = col & 63;
          size_t bh = (size_t)(bb * 8 + hh);
          idx = (omode == 1) ? ((bh << 16) + ((size_t)t << 6) + (size_t)d)
                             : ((bh << 16) + ((size_t)d << 10) + (size_t)t);
        }
        outH[idx] = (_Float16)v;
      }
    }
  }
}

// ---------------------------------------------------------------- attention
// One wave = one (b,h) x 16 query rows; flash-style online softmax over 32-wide
// key blocks.  q,k: f16 [bh][1024][64]; vtr: f16 [bh][64][1024]; o: f16 [8192][512].
__global__ void __launch_bounds__(128) k_attn(
    const _Float16* __restrict__ q, const _Float16* __restrict__ kmat,
    const _Float16* __restrict__ vtr, _Float16* __restrict__ o, float scale) {
  __shared__ __align__(16) float    Ssh[4][512];
  __shared__ __align__(16) _Float16 Psh[4][512];
  __shared__ float Msh[4][16];
  __shared__ float Lsh[4][16];
  __shared__ float Ash[4][16];
  int w = threadIdx.x >> 5;
  int lane = threadIdx.x & 31;
  int gw = blockIdx.x * 4 + w;
  int ttile = gw & 63;   // T/16 = 64 tiles
  int bh = gw >> 6;      // 64 (b,h) pairs
  int t0 = ttile << 4;
  const _Float16* Qb = q + ((size_t)bh << 16);
  const _Float16* Kb = kmat + ((size_t)bh << 16);
  const _Float16* Vb = vtr + ((size_t)bh << 16);
  int lrw = lane & 15;
  int khi = (lane & 16) ? 8 : 0;
  int rb = (lane & 16) ? 8 : 0;

  v16h a0 = frag_load(Qb + (size_t)(t0 + lrw) * 64 + khi);
  v16h a1 = frag_load(Qb + (size_t)(t0 + lrw) * 64 + 32 + khi);
  v8f oacc[4] = {};
  if (lane < 16) { Msh[w][lane] = -1e30f; Lsh[w][lane] = 0.0f; }

  int nsb = (t0 + 15) >> 5;  // last causal 32-block (inclusive), wave-uniform
  for (int sb = 0; sb <= nsb; ++sb) {
    int s0 = sb << 5;
    // S[16,32] = Q[16,64] * K^T
#pragma unroll
    for (int nt = 0; nt < 2; ++nt) {
      v16h b0 = frag_load(Kb + (size_t)(s0 + nt * 16 + lrw) * 64 + khi);
      v16h b1 = frag_load(Kb + (size_t)(s0 + nt * 16 + lrw) * 64 + 32 + khi);
      v8f sc = {};
      sc = wmma_f16f32(a0, b0, sc);
      sc = wmma_f16f32(a1, b1, sc);
      int scol = s0 + nt * 16 + lrw;
#pragma unroll
      for (int r = 0; r < 8; ++r) {
        float v = sc[r] * scale;
        if (scol > t0 + rb + r) v = -1e30f;  // causal mask
        Ssh[w][(rb + r) * 32 + nt * 16 + lrw] = v;
      }
    }
    lds_fence();
    // online softmax: lanes 0..15 own one query row each
    if (lane < 16) {
      float mo = Msh[w][lane];
      float lo = Lsh[w][lane];
      float mx = mo;
      for (int j = 0; j < 32; ++j) mx = fmaxf(mx, Ssh[w][lane * 32 + j]);
      float al = __expf(mo - mx);
      float sum = 0.0f;
      for (int j = 0; j < 32; ++j) {
        float pv = __expf(Ssh[w][lane * 32 + j] - mx);
        Psh[w][lane * 32 + j] = (_Float16)pv;
        sum += pv;
      }
      Msh[w][lane] = mx;
      Lsh[w][lane] = lo * al + sum;
      Ash[w][lane] = al;
    }
    lds_fence();
    float al8[8];
#pragma unroll
    for (int r = 0; r < 8; ++r) al8[r] = Ash[w][rb + r];
#pragma unroll
    for (int n = 0; n < 4; ++n)
#pragma unroll
      for (int r = 0; r < 8; ++r) oacc[n][r] *= al8[r];
    // O[16,64] += P[16,32] * V[32,64]; load P + all 4 V-frags before the WMMAs
    v16h pf = frag_load((const _Float16*)&Psh[w][0] + (size_t)lrw * 32 + khi);
    v16h v0 = frag_load(Vb + (size_t)(0 * 16 + lrw) * 1024 + s0 + khi);
    v16h v1 = frag_load(Vb + (size_t)(1 * 16 + lrw) * 1024 + s0 + khi);
    v16h v2 = frag_load(Vb + (size_t)(2 * 16 + lrw) * 1024 + s0 + khi);
    v16h v3 = frag_load(Vb + (size_t)(3 * 16 + lrw) * 1024 + s0 + khi);
    oacc[0] = wmma_f16f32(pf, v0, oacc[0]);
    oacc[1] = wmma_f16f32(pf, v1, oacc[1]);
    oacc[2] = wmma_f16f32(pf, v2, oacc[2]);
    oacc[3] = wmma_f16f32(pf, v3, oacc[3]);
  }
  if (lane < 16) Ash[w][lane] = 1.0f / Lsh[w][lane];
  lds_fence();
  float il8[8];
#pragma unroll
  for (int r = 0; r < 8; ++r) il8[r] = Ash[w][rb + r];
  int bb = bh >> 3;
  int hh = bh & 7;
#pragma unroll
  for (int n = 0; n < 4; ++n) {
    int col = (hh << 6) + (n << 4) + lrw;
#pragma unroll
    for (int r = 0; r < 8; ++r) {
      int trow = (bb << 10) + t0 + rb + r;
      o[((size_t)trow << 9) + col] = (_Float16)(oacc[n][r] * il8[r]);
    }
  }
}

// ---------------------------------------------------------------- layernorm
// out = LN(x + y) * g + be; one wave per 256-wide row, wave32 shfl reductions.
__global__ void __launch_bounds__(256) k_lnres(
    const float* __restrict__ x, const float* __restrict__ y,
    const float* __restrict__ g, const float* __restrict__ be,
    float* __restrict__ outF, _Float16* __restrict__ outH) {
  int row = blockIdx.x * 8 + (threadIdx.x >> 5);
  int lane = threadIdx.x & 31;
  const float* px = x + (size_t)row * 256;
  const float* py = y + (size_t)row * 256;
  float v[8];
  float s = 0.0f;
#pragma unroll
  for (int i = 0; i < 8; ++i) {
    int c = i * 32 + lane;
    v[i] = px[c] + py[c];
    s += v[i];
  }
#pragma unroll
  for (int m = 16; m >= 1; m >>= 1) s += __shfl_xor(s, m, 32);
  float mean = s * (1.0f / 256.0f);
  float var = 0.0f;
#pragma unroll
  for (int i = 0; i < 8; ++i) { float d = v[i] - mean; var += d * d; }
#pragma unroll
  for (int m = 16; m >= 1; m >>= 1) var += __shfl_xor(var, m, 32);
  var *= (1.0f / 256.0f);
  float inv = rsqrtf(var + 1e-3f);
#pragma unroll
  for (int i = 0; i < 8; ++i) {
    int c = i * 32 + lane;
    float o = (v[i] - mean) * inv * g[c] + be[c];
    outF[(size_t)row * 256 + c] = o;
    outH[(size_t)row * 256 + c] = (_Float16)o;
  }
}

// ---------------------------------------------------------------- converters
__global__ void k_cvt_pad(const float* __restrict__ src, _Float16* __restrict__ dst,
                          int M, int Ks, int Kd) {
  int i = blockIdx.x * 256 + threadIdx.x;
  if (i >= M * Kd) return;
  int r = i / Kd, c = i - r * Kd;
  dst[i] = (c < Ks) ? (_Float16)src[(size_t)r * Ks + c] : (_Float16)0.0f;
}

// src f32 [K][N] -> dst f16 [N][K]
__global__ void k_transpose_w(const float* __restrict__ src, _Float16* __restrict__ dst,
                              int K, int N) {
  int i = blockIdx.x * 256 + threadIdx.x;
  if (i >= K * N) return;
  int n = i / K, k = i - n * K;
  dst[i] = (_Float16)src[(size_t)k * N + n];
}

// conv weight tap: src f32 [O][C][3] -> dst f16 [O][Cpad], dst[o][c] = src[o][c][tap]
__global__ void k_convw(const float* __restrict__ src, _Float16* __restrict__ dst,
                        int O, int C, int Cpad, int tap) {
  int i = blockIdx.x * 256 + threadIdx.x;
  if (i >= O * Cpad) return;
  int o = i / Cpad, c = i - o * Cpad;
  dst[i] = (c < C) ? (_Float16)src[((size_t)o * C + c) * 3 + tap] : (_Float16)0.0f;
}

__global__ void k_zero16(_Float16* __restrict__ p, int n) {
  int i = blockIdx.x * 256 + threadIdx.x;
  if (i < n) p[i] = (_Float16)0.0f;
}

// ---------------------------------------------------------------- host
static inline int cdiv_h(int a, int b) { return (a + b - 1) / b; }

extern "C" void kernel_launch(void* const* d_in, const int* in_sizes, int n_in,
                              void* d_out, int out_size, void* d_ws, size_t ws_size,
                              hipStream_t stream) {
  (void)in_sizes; (void)n_in; (void)out_size; (void)ws_size;
  const float* x = (const float*)d_in[0];
  const float* convw[4] = {(const float*)d_in[1], (const float*)d_in[2],
                           (const float*)d_in[3], (const float*)d_in[4]};
  const float* convb[4] = {(const float*)d_in[5], (const float*)d_in[6],
                           (const float*)d_in[7], (const float*)d_in[8]};
  const float* Wq  = (const float*)d_in[9];
  const float* bq  = (const float*)d_in[10];
  const float* Wk  = (const float*)d_in[11];
  const float* bk  = (const float*)d_in[12];
  const float* Wv  = (const float*)d_in[13];
  const float* bv  = (const float*)d_in[14];
  const float* Wo1 = (const float*)d_in[15];
  const float* bo1 = (const float*)d_in[16];
  const float* g1  = (const float*)d_in[17];
  const float* be1 = (const float*)d_in[18];
  const float* Wi  = (const float*)d_in[19];
  const float* bi  = (const float*)d_in[20];
  const float* g2  = (const float*)d_in[21];
  const float* be2 = (const float*)d_in[22];
  const float* Wl  = (const float*)d_in[23];
  const float* bl  = (const float*)d_in[24];
  float* out = (float*)d_out;

  // ---- workspace carve-up (all offsets 256B aligned)
  char* base = (char*)d_ws;
  size_t off = 0;
  auto take = [&](size_t bytes) -> char* {
    char* r = base + off;
    off = (off + bytes + 255) & ~(size_t)255;
    return r;
  };
  _Float16* zbuf = (_Float16*)take(1056ull * 2);  // zero page for OOB lanes
  float*    hF  = (float*)take(8192ull * 256 * 4);
  _Float16* hH  = (_Float16*)take(8192ull * 256 * 2);
  float*    h1F = (float*)take(8192ull * 256 * 4);
  _Float16* h1H = (_Float16*)take(8192ull * 256 * 2);
  float*    dF  = (float*)take(8192ull * 256 * 4);
  _Float16* wqT = (_Float16*)take(512ull * 256 * 2);
  _Float16* wkT = (_Float16*)take(512ull * 256 * 2);
  _Float16* wvT = (_Float16*)take(512ull * 256 * 2);
  _Float16* wiT = (_Float16*)take(512ull * 256 * 2);
  _Float16* woT = (_Float16*)take(256ull * 512 * 2);
  _Float16* wlT = (_Float16*)take(257ull * 256 * 2);
  _Float16* wt[3];
  for (int t = 0; t < 3; ++t) wt[t] = (_Float16*)take(1024ull * 1024 * 2);
  char* region = take(66ull << 20);  // shared big region, reused across phases
  // conv phase layout
  _Float16* xh      = (_Float16*)(region);
  _Float16* act1    = (_Float16*)(region + 4718592);
  _Float16* act2    = (_Float16*)(region + 4718592 + 16777216);
  _Float16* act3    = (_Float16*)(region + 4718592 + 16777216 + 8388608);
  float*    convacc = (float*)(region + 4718592 + 16777216 + 8388608 + 2097152);
  // transformer phase layout (conv buffers are dead by then)
  _Float16* qH   = (_Float16*)(region);
  _Float16* kH   = (_Float16*)(region + 1ull * 8388608);
  _Float16* vtH  = (_Float16*)(region + 2ull * 8388608);
  _Float16* oH   = (_Float16*)(region + 3ull * 8388608);
  _Float16* intH = (_Float16*)(region + 4ull * 8388608);

  auto gemm = [&](const _Float16* A, int lda, int joff, const _Float16* Bt, int ldb,
                  const float* bias, const float* accsrc, float* oF, _Float16* oHp,
                  int N, int K, int act, int omode) {
    int tiles = 512 * cdiv_h(N, 64);  // M = 8192 -> 512 row tiles, 64-wide N blocks
    k_gemm<<<cdiv_h(tiles, 8), 256, 0, stream>>>(A, lda, joff, Bt, ldb, bias, accsrc,
                                                 oF, oHp, zbuf, 8192, N, K, act, omode);
  };

  // zero page first (ws is poisoned with 0xAA by the harness)
  k_zero16<<<5, 256, 0, stream>>>(zbuf, 1056);

  // ---- conv stem:  x [8192,257] -> h [8192,256], each layer = 3 shifted GEMMs
  k_cvt_pad<<<cdiv_h(8192 * 288, 256), 256, 0, stream>>>(x, xh, 8192, 257, 288);
  const int cO[4] = {1024, 512, 128, 256};
  const int cC[4] = {257, 1024, 512, 128};
  const int cP[4] = {288, 1024, 512, 128};  // padded K (multiple of 32)
  const _Float16* cA[4] = {xh, act1, act2, act3};
  _Float16* cOut[3] = {act1, act2, act3};
  for (int l = 0; l < 4; ++l) {
    for (int t = 0; t < 3; ++t)
      k_convw<<<cdiv_h(cO[l] * cP[l], 256), 256, 0, stream>>>(convw[l], wt[t], cO[l],
                                                              cC[l], cP[l], t);
    for (int t = 0; t < 3; ++t) {
      bool last = (t == 2);
      const float* accs = (t > 0) ? convacc : nullptr;
      float* oF = nullptr;
      _Float16* oHp = nullptr;
      const float* bptr = nullptr;
      int act = 0;
      if (!last) {
        oF = convacc;  // partial sums (pass0 overwrites poisoned ws)
      } else {
        bptr = convb[l];
        act = 1;  // bias + leaky relu on final tap
        if (l < 3) oHp = cOut[l];
        else { oF = hF; oHp = hH; }
      }
      gemm(cA[l], cP[l], t - 2, wt[t], cP[l], bptr, accs, oF, oHp, cO[l], cP[l], act, 0);
    }
  }

  // ---- transformer layers
  const float ATTN_SCALE = 0.04419417382415922f;  // 512^-0.5 (faithful to source)
  for (int i = 0; i < 8; ++i) {
    k_transpose_w<<<512, 256, 0, stream>>>(Wq + (size_t)i * 131072, wqT, 256, 512);
    k_transpose_w<<<512, 256, 0, stream>>>(Wk + (size_t)i * 131072, wkT, 256, 512);
    k_transpose_w<<<512, 256, 0, stream>>>(Wv + (size_t)i * 131072, wvT, 256, 512);
    k_transpose_w<<<512, 256, 0, stream>>>(Wo1 + (size_t)i * 131072, woT, 512, 256);
    k_transpose_w<<<512, 256, 0, stream>>>(Wi + (size_t)i * 131072, wiT, 256, 512);
    // q,k -> [b,h,t,d]; v -> [b,h,d,t] (pre-transposed for B-fragments)
    gemm(hH, 256, 0, wqT, 256, bq + i * 512, nullptr, nullptr, qH, 512, 256, 0, 1);
    gemm(hH, 256, 0, wkT, 256, bk + i * 512, nullptr, nullptr, kH, 512, 256, 0, 1);
    gemm(hH, 256, 0, wvT, 256, bv + i * 512, nullptr, nullptr, vtH, 512, 256, 0, 2);
    k_attn<<<1024, 128, 0, stream>>>(qH, kH, vtH, oH, ATTN_SCALE);
    // output1 projection + residual + LN
    gemm(oH, 512, 0, woT, 512, bo1 + i * 256, nullptr, dF, nullptr, 256, 512, 0, 0);
    k_lnres<<<1024, 256, 0, stream>>>(hF, dF, g1 + i * 256, be1 + i * 256, h1F, h1H);
    // FFN (second projection faithfully reuses Wo1/bo1 per source)
    gemm(h1H, 256, 0, wiT, 256, bi + i * 512, nullptr, nullptr, intH, 512, 256, 1, 0);
    gemm(intH, 512, 0, woT, 512, bo1 + i * 256, nullptr, dF, nullptr, 256, 512, 0, 0);
    k_lnres<<<1024, 256, 0, stream>>>(h1F, dF, g2 + i * 256, be2 + i * 256, hF, hH);
  }

  // ---- final linear + relu -> d_out [8192, 257] fp32
  k_transpose_w<<<cdiv_h(257 * 256, 256), 256, 0, stream>>>(Wl, wlT, 256, 257);
  gemm(hH, 256, 0, wlT, 256, bl, nullptr, out, nullptr, 257, 256, 2, 0);
}
